// ETP_55499567398928
// MI455X (gfx1250) — matrix operand order repaired and checked by reading, not compiled, not run on hardware
//
#include <hip/hip_runtime.h>
#include <math.h>

// ---------------- problem constants (from reference) ----------------
#define N_ROWS 4096      // n
#define N_COLS 8192      // m
#define ALPHA  0.1f
#define EPS_F  1e-9f
#define STOPTHR 1e-9f
#define TP_ELEMS 8
#define TP_BLOCKS ((int)(((size_t)N_ROWS * N_COLS) / (256 * TP_ELEMS)))  // 16384

typedef _Float16 h8  __attribute__((ext_vector_type(8)));
typedef _Float16 h16 __attribute__((ext_vector_type(16)));
typedef float    f8  __attribute__((ext_vector_type(8)));

// ---------------- init: u0 = 1/n (uhat = 1), v = 0, flags reset ----------------
__global__ __launch_bounds__(256)
void init_kernel(float* __restrict__ u, _Float16* __restrict__ uhat,
                 float* __restrict__ v, _Float16* __restrict__ vhat,
                 float* __restrict__ err2, int* __restrict__ done)
{
    int i = blockIdx.x * blockDim.x + threadIdx.x;
    if (i < N_ROWS) { u[i] = 1.0f / N_ROWS; uhat[i] = (_Float16)1.0f; }
    if (i < N_COLS) { v[i] = 0.0f;          vhat[i] = (_Float16)0.0f; }
    if (i == 0) { err2[0] = 0.0f; err2[1] = 0.0f; *done = 0; }
}

// ---------------- K = exp(-alpha*M) -> f16, row-major + transposed ----------------
__global__ __launch_bounds__(256)
void kexp_kernel(const float* __restrict__ M,
                 _Float16* __restrict__ K16, _Float16* __restrict__ K16T)
{
    __shared__ _Float16 tile[32][33];
    const int tx = threadIdx.x;            // 0..31
    const int ty = threadIdx.y;            // 0..7
    const int col0 = blockIdx.x * 32;
    const int row0 = blockIdx.y * 32;
    for (int rr = ty; rr < 32; rr += 8) {
        float m = M[(size_t)(row0 + rr) * N_COLS + col0 + tx];
        _Float16 k = (_Float16)expf(-ALPHA * m);
        K16[(size_t)(row0 + rr) * N_COLS + col0 + tx] = k;
        tile[rr][tx] = k;
    }
    __syncthreads();
    for (int rr = ty; rr < 32; rr += 8) {
        K16T[(size_t)(col0 + rr) * N_ROWS + row0 + tx] = tile[tx][rr];
    }
}

// ---------------- WMMA GEMV core: 16 rows of W[R][C] (f16) dot xhat (f16) ----------------
// Each of 4 waves takes C/4 of the k-range; D column 0 holds the 16 partial dots.
// 16-bit A 16x32 layout: lane<16 holds K 0..7 / 16..23; lane>=16 holds K 8..15 / 24..31.
// 16-bit B 32x16 layout: lane<16 = one column with K 0..7 / 16..23; lane>=16 = K 8..15 / 24..31.
__device__ inline f8 wmma_dot16(const _Float16* __restrict__ W,
                                const _Float16* __restrict__ xhat,
                                int rowBase, int C)
{
    const int lane = threadIdx.x & 31;
    const int wave = threadIdx.x >> 5;
    const int kChunk = C >> 2;                 // per-wave k-range (multiple of 32)
    const int k0w = wave * kChunk;
    const int r = rowBase + (lane & 15);
    const int koff = (lane & 16) ? 8 : 0;
    const int xsel = (lane & 16) ? 8 : 0;
    const _Float16* __restrict__ Arow = W + (size_t)r * C;

    f8 acc = {};
    for (int k = 0; k < kChunk; k += 32) {
        const int kk = k0w + k;
        __builtin_prefetch(Arow + kk + 128, 0, 0);   // global_prefetch_b8, 4 iters ahead
        h8 alo = *(const h8*)(Arow + kk + koff);
        h8 ahi = *(const h8*)(Arow + kk + koff + 16);
        h8 blo = *(const h8*)(xhat + kk + xsel);
        h8 bhi = *(const h8*)(xhat + kk + xsel + 16);
        h16 a = __builtin_shufflevector(alo, ahi, 0,1,2,3,4,5,6,7,8,9,10,11,12,13,14,15);
        h16 b = __builtin_shufflevector(blo, bhi, 0,1,2,3,4,5,6,7,8,9,10,11,12,13,14,15);
        acc = __builtin_amdgcn_wmma_f32_16x16x32_f16(
            /*neg_a=*/false, a, /*neg_b=*/false, b,
            /*c_mod=*/(short)0, acc, /*reuse_a=*/false, /*reuse_b=*/false);
    }
    return acc;
}

// D(16x16 f32) layout: VGPR g, lanes0-15 -> M=g,N=lane ; lanes16-31 -> M=8+g,N=lane-16.
// Column N=0 lives in lane 0 (rows 0..7) and lane 16 (rows 8..15).
__device__ inline void store_col0(float (*red)[16], const f8& acc)
{
    const int lane = threadIdx.x & 31;
    const int wave = threadIdx.x >> 5;
    if (lane == 0) {
        #pragma unroll
        for (int g = 0; g < 8; ++g) red[wave][g] = acc[g];
    } else if (lane == 16) {
        #pragma unroll
        for (int g = 0; g < 8; ++g) red[wave][8 + g] = acc[g];
    }
}

// ---------------- one Sinkhorn half-step: y = invR / (dot*invC + eps), yhat = 1/(..) ----------------
__global__ __launch_bounds__(128)
void gemv_wmma_kernel(const _Float16* __restrict__ W,     // [R][C] f16 row-major
                      const _Float16* __restrict__ xhat,  // [C] scaled f16
                      float* __restrict__ y,              // [R] true marginal vector
                      _Float16* __restrict__ yhat,        // [R] scaled f16
                      int C, float invC, float invR,
                      const int* __restrict__ done)
{
    if (*(volatile const int*)done) return;
    __shared__ float red[4][16];
    const int rowBase = blockIdx.x * 16;
    f8 acc = wmma_dot16(W, xhat, rowBase, C);
    store_col0(red, acc);
    __syncthreads();
    if (threadIdx.x < 16) {
        const int t = threadIdx.x;
        float d = red[0][t] + red[1][t] + red[2][t] + red[3][t];
        float denom = d * invC + EPS_F;
        y[rowBase + t]    = invR / denom;
        yhat[rowBase + t] = (_Float16)(1.0f / denom);
    }
}

// ---------------- err = sum_j | v_j * (K^T u)_j - 1/m | ----------------
__global__ __launch_bounds__(128)
void err_wmma_kernel(const _Float16* __restrict__ Wt,     // K16T [V][N]
                     const _Float16* __restrict__ uhat,   // [N] scaled
                     const float* __restrict__ v,         // [V] true v
                     float* __restrict__ err_slot,
                     const int* __restrict__ done)
{
    if (*(volatile const int*)done) return;
    __shared__ float red[4][16];
    __shared__ float esum[16];
    const int rowBase = blockIdx.x * 16;
    f8 acc = wmma_dot16(Wt, uhat, rowBase, N_ROWS);
    store_col0(red, acc);
    __syncthreads();
    if (threadIdx.x < 16) {
        const int t = threadIdx.x;
        float d = red[0][t] + red[1][t] + red[2][t] + red[3][t];
        float bb = v[rowBase + t] * (d * (1.0f / N_ROWS));
        esum[t] = fabsf(bb - 1.0f / N_COLS);
    }
    __syncthreads();
    if (threadIdx.x == 0) {
        float s = 0.0f;
        #pragma unroll
        for (int t = 0; t < 16; ++t) s += esum[t];
        atomicAdd(err_slot, s);
    }
}

__global__ void finalize_kernel(const float* __restrict__ err_slot, int* __restrict__ done)
{
    if (threadIdx.x == 0 && blockIdx.x == 0) {
        if (*err_slot <= STOPTHR) *done = 1;
    }
}

// ---------------- transp = u * exp(-a*M) * v^T ; loss partials ----------------
__global__ __launch_bounds__(256)
void transp_kernel(const float* __restrict__ M, const float* __restrict__ u,
                   const float* __restrict__ v, float* __restrict__ out,
                   float* __restrict__ partial)
{
    __shared__ float sh[256];
    const size_t idx0 = ((size_t)blockIdx.x * 256 + threadIdx.x) * TP_ELEMS;
    const int row = (int)(idx0 >> 13);       // / 8192
    const int col = (int)(idx0 & (N_COLS - 1));
    const float uu = u[row];
    float s = 0.0f;
    #pragma unroll
    for (int e = 0; e < TP_ELEMS; ++e) {
        float m = M[idx0 + e];
        float t = uu * expf(-ALPHA * m) * v[col + e];
        out[1 + idx0 + e] = t;
        s += t * m;
    }
    sh[threadIdx.x] = s;
    __syncthreads();
    for (int o = 128; o > 0; o >>= 1) {
        if (threadIdx.x < (unsigned)o) sh[threadIdx.x] += sh[threadIdx.x + o];
        __syncthreads();
    }
    if (threadIdx.x == 0) partial[blockIdx.x] = sh[0];
}

__global__ __launch_bounds__(256)
void reduce_kernel(const float* __restrict__ partial, int n, float* __restrict__ out0)
{
    __shared__ float sh[256];
    float s = 0.0f;
    for (int i = threadIdx.x; i < n; i += 256) s += partial[i];
    sh[threadIdx.x] = s;
    __syncthreads();
    for (int o = 128; o > 0; o >>= 1) {
        if (threadIdx.x < (unsigned)o) sh[threadIdx.x] += sh[threadIdx.x + o];
        __syncthreads();
    }
    if (threadIdx.x == 0) out0[0] = sh[0];
}

// ---------------- host side ----------------
extern "C" void kernel_launch(void* const* d_in, const int* in_sizes, int n_in,
                              void* d_out, int out_size, void* d_ws, size_t ws_size,
                              hipStream_t stream)
{
    (void)in_sizes; (void)n_in; (void)out_size; (void)ws_size;
    const float* M = (const float*)d_in[0];
    float* out = (float*)d_out;

    char* w = (char*)d_ws;
    size_t off = 0;
    auto alloc = [&](size_t bytes) -> char* {
        off = (off + 255) & ~(size_t)255;
        char* p = w + off;
        off += bytes;
        return p;
    };
    const size_t NV = (size_t)N_ROWS * N_COLS;
    _Float16* K16  = (_Float16*)alloc(NV * 2);          // K row-major f16   (67 MB, L2-resident)
    _Float16* K16T = (_Float16*)alloc(NV * 2);          // K^T row-major f16 (67 MB, L2-resident)
    float*    u    = (float*)   alloc(N_ROWS * 4);
    float*    v    = (float*)   alloc(N_COLS * 4);
    _Float16* uhat = (_Float16*)alloc(N_ROWS * 2);      // u * n  (O(1), f16)
    _Float16* vhat = (_Float16*)alloc(N_COLS * 2);      // v * m  (O(1), f16)
    float*    err2 = (float*)   alloc(2 * 4);
    int*      done = (int*)     alloc(4);
    float*    part = (float*)   alloc((size_t)TP_BLOCKS * 4);

    init_kernel<<<(N_COLS + 255) / 256, 256, 0, stream>>>(u, uhat, v, vhat, err2, done);
    kexp_kernel<<<dim3(N_COLS / 32, N_ROWS / 32), dim3(32, 8), 0, stream>>>(M, K16, K16T);

    for (int it = 0; it < 100; ++it) {
        // v = b / (K^T u + eps)   (scaled: vhat = 1/(dot/n + eps))
        gemv_wmma_kernel<<<N_COLS / 16, 128, 0, stream>>>(
            K16T, uhat, v, vhat, N_ROWS, 1.0f / N_ROWS, 1.0f / N_COLS, done);
        // u = a / (K v + eps)
        gemv_wmma_kernel<<<N_ROWS / 16, 128, 0, stream>>>(
            K16, vhat, u, uhat, N_COLS, 1.0f / N_COLS, 1.0f / N_ROWS, done);
        if (it == 0 || it == 50) {          // cpt % 50 == 1
            float* slot = err2 + (it == 0 ? 0 : 1);
            err_wmma_kernel<<<N_COLS / 16, 128, 0, stream>>>(K16T, uhat, v, slot, done);
            finalize_kernel<<<1, 32, 0, stream>>>(slot, done);
        }
    }

    transp_kernel<<<TP_BLOCKS, 256, 0, stream>>>(M, u, v, out, part);
    reduce_kernel<<<1, 256, 0, stream>>>(part, TP_BLOCKS, out);
}
